// SetTransformerConditionEncoder_22565758174077
// MI455X (gfx1250) — matrix operand order repaired and checked by reading, not compiled, not run on hardware
//
#include <hip/hip_runtime.h>
#include <hip/hip_bf16.h>
#include <math.h>

// ---------------------------------------------------------------------------
// CDNA5 (gfx1250) wave32 WMMA types & fragment helpers
// ---------------------------------------------------------------------------
typedef __attribute__((ext_vector_type(16))) _Float16 v16h;
typedef __attribute__((ext_vector_type(8)))  _Float16 v8h;
typedef __attribute__((ext_vector_type(8)))  float    v8f;

#define SCALE_QK 0.044194173824159216f   // 1/sqrt(512)

__device__ __forceinline__ v8f zero8() {
  v8f r;
#pragma unroll
  for (int i = 0; i < 8; ++i) r[i] = 0.f;
  return r;
}

__device__ __forceinline__ v8f wmma16(v16h a, v16h b, v8f c) {
  return __builtin_amdgcn_wmma_f32_16x16x32_f16(false, a, false, b, (short)0, c, false, false);
}

__device__ __forceinline__ unsigned pack2h(float a, float b) {
  union { _Float16 h[2]; unsigned u; } c;
  c.h[0] = (_Float16)a; c.h[1] = (_Float16)b;
  return c.u;
}

// A fragment 16x32 (MxK) from f16 row-major source, leading dim ld (halves).
// Lane L: m=L&15, hf=L>>4; pair g: K = 2*(g&3) + 16*(g>>2) + 8*hf.
__device__ __forceinline__ v16h a_frag16(const _Float16* src, int ld) {
  int l = threadIdx.x & 31, m = l & 15, hf = l >> 4;
  const _Float16* row = src + (size_t)m * ld;
  v16h a;
#pragma unroll
  for (int g = 0; g < 8; ++g) {
    int k = ((g & 3) << 1) + ((g >> 2) << 4) + (hf << 3);
    union { unsigned u; _Float16 h[2]; } c;
    c.u = *(const unsigned*)(row + k);
    a[2 * g] = c.h[0]; a[2 * g + 1] = c.h[1];
  }
  return a;
}

// B fragment 32x16 (KxN) from f16 K-contiguous storage: element(K) = base[n*ldk + K].
// Two 16-byte loads per lane, no converts.
__device__ __forceinline__ v16h b_fragW(const _Float16* base, int ldk) {
  int l = threadIdx.x & 31, n = l & 15, hf = l >> 4;
  const _Float16* p = base + (size_t)n * ldk + 8 * hf;
  v8h lo = *(const v8h*)p;
  v8h hi = *(const v8h*)(p + 16);
  v16h b;
#pragma unroll
  for (int e = 0; e < 8; ++e) { b[e] = lo[e]; b[e + 8] = hi[e]; }
  return b;
}

// ---------------------------------------------------------------------------
// 0) Weight pre-pass: WT[n][k] = (f16) W[k][n]   (512x512)
// ---------------------------------------------------------------------------
__global__ void wtrans_kernel(const float* __restrict__ W, _Float16* __restrict__ WT) {
  __shared__ float tile[32][33];
  int bx = blockIdx.x & 15, by = blockIdx.x >> 4;      // n-block, k-block
  int tx = threadIdx.x & 31, ty = threadIdx.x >> 5;    // 32 x 8
#pragma unroll
  for (int i = 0; i < 32; i += 8)
    tile[ty + i][tx] = W[(size_t)(by * 32 + ty + i) * 512 + bx * 32 + tx];
  __syncthreads();
#pragma unroll
  for (int i = 0; i < 32; i += 8)
    WT[(size_t)(bx * 32 + ty + i) * 512 + by * 32 + tx] = (_Float16)tile[tx][ty + i];
}

// ---------------------------------------------------------------------------
// 1) Input embedding: x = gelu(concat(xv, res) @ Wi + bi) * 0.1
// ---------------------------------------------------------------------------
__global__ void embed_kernel(const float* __restrict__ xv, const float* __restrict__ res,
                             const float* __restrict__ Wi, const float* __restrict__ bi,
                             float* __restrict__ x) {
  size_t row = blockIdx.x;
  int t = threadIdx.x;
  float f[7];
  const float* xr = xv + row * 6;
#pragma unroll
  for (int j = 0; j < 6; ++j) f[j] = xr[j];
  f[6] = res[row];
#pragma unroll
  for (int i = 0; i < 2; ++i) {
    int c = t + 256 * i;
    float acc = bi[c];
#pragma unroll
    for (int j = 0; j < 7; ++j) acc += f[j] * Wi[j * 512 + c];
    float g = 0.5f * acc * (1.f + erff(acc * 0.70710678118654752f));
    x[row * 512 + c] = g * 0.1f;
  }
}

// ---------------------------------------------------------------------------
// 2) Generic WMMA GEMM: C[Mx512] = A[Mx512] @ W[512x512] + bias
//    OUT_MODE: 0 = f32 row-major; 1 = f16 row-major; 2 = f16 [b][feat][key]
// ---------------------------------------------------------------------------
template <int OUT_MODE>
__global__ void gemm_bias_512(const float* __restrict__ A, int lda,
                              const _Float16* __restrict__ WT,
                              const float* __restrict__ bias,
                              void* __restrict__ Cout, int ldc, int Mvalid) {
  __shared__ _Float16 at[16 * 512];     // 16 KB
  int tile = blockIdx.x;
  int wave = threadIdx.x >> 5;
  int l = threadIdx.x & 31, n0 = l & 15, hf = l >> 4;
  int mval = Mvalid - tile * 16;

  unsigned* atu = (unsigned*)at;
  for (int i4 = threadIdx.x; i4 < 16 * 128; i4 += 256) {
    int m = i4 >> 7, c4 = (i4 & 127) << 2;
    float4 vv; vv.x = vv.y = vv.z = vv.w = 0.f;
    if (m < mval) vv = *(const float4*)(A + (size_t)(tile * 16 + m) * lda + c4);
    int o = (m * 512 + c4) >> 1;
    atu[o] = pack2h(vv.x, vv.y);
    atu[o + 1] = pack2h(vv.z, vv.w);
  }
  __syncthreads();

  v8f acc[4] = {zero8(), zero8(), zero8(), zero8()};
  for (int k = 0; k < 512; k += 32) {
    v16h a = a_frag16(at + k, 512);
#pragma unroll
    for (int j = 0; j < 4; ++j) {
      int nf = wave + 8 * j;
      v16h b = b_fragW(WT + (size_t)(nf * 16) * 512 + k, 512);
      acc[j] = wmma16(a, b, acc[j]);
    }
  }
#pragma unroll
  for (int j = 0; j < 4; ++j) {
    int nf = wave + 8 * j;
#pragma unroll
    for (int r = 0; r < 8; ++r) {
      int m = r + 8 * hf, n = nf * 16 + n0;
      int row = tile * 16 + m;
      float val = acc[j][r] + bias[n];
      if (OUT_MODE == 0) {
        ((float*)Cout)[(size_t)row * ldc + n] = val;
      } else if (OUT_MODE == 1) {
        ((_Float16*)Cout)[(size_t)row * ldc + n] = (_Float16)val;
      } else {
        int bb = row >> 6, key = row & 63;
        ((_Float16*)Cout)[((size_t)bb * 512 + n) * 64 + key] = (_Float16)val;
      }
    }
  }
}

// ---------------------------------------------------------------------------
// 3) Flash attention for MAB0 / PMA: out = q + softmax(q K^T/sqrt(D)) V
// ---------------------------------------------------------------------------
template <int MQ>
__global__ void isab_attn(const float* __restrict__ x, const float* __restrict__ q,
                          const _Float16* __restrict__ WkT, const float* __restrict__ bk,
                          const _Float16* __restrict__ WvT, const float* __restrict__ bv,
                          float* __restrict__ out, int nq_store) {
  constexpr int SF = (MQ / 16) * 4;
  constexpr int MAXF = (SF + 7) / 8;
  int b = blockIdx.x >> 3;
  int h = blockIdx.x & 7;
  int wave = threadIdx.x >> 5;
  int l = threadIdx.x & 31, n0 = l & 15, hf = l >> 4;
  int tid = threadIdx.x;

  __shared__ _Float16 kt16[64 * 64];    // [key][feat]   8 KB
  __shared__ _Float16 vtT[64 * 64];     // [feat][key]   8 KB
  __shared__ float scxs[64 * 64];       // scores f32 / x-slice f16 (16 KB, shared)
  __shared__ _Float16 qs[MQ * 64];      // q head slice f16
  __shared__ float mrow[MQ], lrow[MQ], frow[MQ];

  float* sc = scxs;                               // MQ x 64 f32
  _Float16* xs = (_Float16*)scxs;                 // 64 x 128 f16
  _Float16* psc = (_Float16*)scxs;                // P rows, stride 128 halves

  const float* xb  = x + (size_t)b * 4096 * 512;
  const _Float16* WkTh = WkT + (size_t)(h * 64) * 512;
  const _Float16* WvTh = WvT + (size_t)(h * 64) * 512;
  const float* qh  = q + h * 64;

  for (int i4 = tid; i4 < MQ * 16; i4 += 256) {
    int m = i4 >> 4, c4 = (i4 & 15) << 2;
    float4 vv = *(const float4*)(qh + (size_t)m * 512 + c4);
    unsigned* d = (unsigned*)(qs + m * 64 + c4);
    d[0] = pack2h(vv.x, vv.y);
    d[1] = pack2h(vv.z, vv.w);
  }
  for (int i = tid; i < MQ; i += 256) { mrow[i] = -3.0e38f; lrow[i] = 0.f; }

  v8f oacc[MAXF];
#pragma unroll
  for (int i = 0; i < MAXF; ++i) oacc[i] = zero8();

  for (int t = 0; t < 4096; t += 64) {
    // ---- k_tile/v_tile (64x64): 4 frags per wave, K staged in 128-wide slices
    v8f kv[4] = {zero8(), zero8(), zero8(), zero8()};
    for (int ks = 0; ks < 512; ks += 128) {
      __syncthreads();   // also protects scxs from previous iteration's readers
      unsigned* xsu = (unsigned*)xs;
      for (int i4 = tid; i4 < 64 * 32; i4 += 256) {
        int m = i4 >> 5, c4 = (i4 & 31) << 2;
        float4 vv = *(const float4*)(xb + (size_t)(t + m) * 512 + ks + c4);
        int o = (m * 128 + c4) >> 1;
        xsu[o] = pack2h(vv.x, vv.y);
        xsu[o + 1] = pack2h(vv.z, vv.w);
      }
      __syncthreads();
#pragma unroll
      for (int j = 0; j < 4; ++j) {
        int f = wave + 8 * j;
        bool isv = f >= 16;
        int ff = f & 15, mt = ff >> 2, nt = ff & 3;
        const _Float16* WTh = isv ? WvTh : WkTh;
#pragma unroll
        for (int k = 0; k < 128; k += 32) {
          v16h a  = a_frag16(xs + (mt * 16) * 128 + k, 128);
          v16h bf = b_fragW(WTh + (size_t)(nt * 16) * 512 + ks + k, 512);
          kv[j] = wmma16(a, bf, kv[j]);
        }
      }
    }
    __syncthreads();
#pragma unroll
    for (int j = 0; j < 4; ++j) {
      int f = wave + 8 * j;
      bool isv = f >= 16;
      int ff = f & 15, mt = ff >> 2, nt = ff & 3;
      const float* bb = isv ? bv : bk;
#pragma unroll
      for (int r = 0; r < 8; ++r) {
        int m = mt * 16 + r + 8 * hf, n = nt * 16 + n0;
        float val = kv[j][r] + bb[h * 64 + n];
        if (isv) vtT[n * 64 + m] = (_Float16)val;
        else     kt16[m * 64 + n] = (_Float16)val;
      }
    }
    __syncthreads();
    // ---- scores = q_h @ k_tile^T * scale
    for (int f = wave; f < SF; f += 8) {
      int mt = f >> 2, nt = f & 3;
      v8f acc = zero8();
#pragma unroll
      for (int kk = 0; kk < 64; kk += 32) {
        v16h a  = a_frag16(qs + (mt * 16) * 64 + kk, 64);
        v16h bf = b_fragW(kt16 + (size_t)(nt * 16) * 64 + kk, 64);
        acc = wmma16(a, bf, acc);
      }
#pragma unroll
      for (int r = 0; r < 8; ++r)
        sc[(mt * 16 + r + 8 * hf) * 64 + nt * 16 + n0] = acc[r] * SCALE_QK;
    }
    __syncthreads();
    // ---- online softmax; write P as f16 in place (row stride 128 halves)
    for (int row = wave; row < MQ; row += 8) {
      float* srow = sc + row * 64;
      float v0 = srow[l], v1 = srow[l + 32];
      float mx = fmaxf(v0, v1);
#pragma unroll
      for (int off = 16; off > 0; off >>= 1) mx = fmaxf(mx, __shfl_xor(mx, off, 32));
      float mold = mrow[row];
      float mnew = fmaxf(mold, mx);
      float p0 = __expf(v0 - mnew), p1 = __expf(v1 - mnew);
      float s = p0 + p1;
#pragma unroll
      for (int off = 16; off > 0; off >>= 1) s += __shfl_xor(s, off, 32);
      _Float16* prow = (_Float16*)srow;
      prow[l] = (_Float16)p0; prow[l + 32] = (_Float16)p1;
      if (l == 0) {
        float fac = __expf(mold - mnew);
        frow[row] = fac;
        lrow[row] = lrow[row] * fac + s;
        mrow[row] = mnew;
      }
    }
    __syncthreads();
    // ---- rescale O and accumulate P @ v_tile
    for (int f = wave, li = 0; f < SF; f += 8, ++li) {
      int mt = f >> 2, nt = f & 3;
#pragma unroll
      for (int r = 0; r < 8; ++r) oacc[li][r] *= frow[mt * 16 + r + 8 * hf];
#pragma unroll
      for (int kk = 0; kk < 64; kk += 32) {
        v16h a  = a_frag16(psc + (mt * 16) * 128 + kk, 128);
        v16h bf = b_fragW(vtT + (size_t)(nt * 16) * 64 + kk, 64);
        oacc[li] = wmma16(a, bf, oacc[li]);
      }
    }
  }
  __syncthreads();
  // ---- finalize: out = q + acc / l
  for (int f = wave, li = 0; f < SF; f += 8, ++li) {
    int mt = f >> 2, nt = f & 3;
#pragma unroll
    for (int r = 0; r < 8; ++r) {
      int m = mt * 16 + r + 8 * hf;
      int n = nt * 16 + n0;
      if (m < nq_store) {
        float val = qh[(size_t)m * 512 + n] + oacc[li][r] / lrow[m];
        out[((size_t)b * nq_store + m) * 512 + h * 64 + n] = val;
      }
    }
  }
}

// ---------------------------------------------------------------------------
// Shared post block: LN0 -> (x + relu(x@Wo + bo)) -> LN1 -> dst
// O16 is a 16x512 f16 LDS scratch: LN0 output mirrored there for the FFN A side.
// ---------------------------------------------------------------------------
__device__ __forceinline__ void post_block(float* Obuf, _Float16* O16,
                                           const _Float16* __restrict__ WoT,
                                           const float* __restrict__ bo,
                                           const float* __restrict__ g0, const float* __restrict__ b0,
                                           const float* __restrict__ g1, const float* __restrict__ b1,
                                           float* dst, int dst_stride, int nrows) {
  int wave = threadIdx.x >> 5, l = threadIdx.x & 31, n0 = l & 15, hf = l >> 4;
  for (int row = wave; row < 16; row += 8) {
    float* r = Obuf + row * 512;
    float s = 0.f;
    for (int c = l; c < 512; c += 32) s += r[c];
#pragma unroll
    for (int off = 16; off > 0; off >>= 1) s += __shfl_xor(s, off, 32);
    float mean = s * (1.f / 512.f);
    float v = 0.f;
    for (int c = l; c < 512; c += 32) { float d = r[c] - mean; v += d * d; }
#pragma unroll
    for (int off = 16; off > 0; off >>= 1) v += __shfl_xor(v, off, 32);
    float rstd = rsqrtf(v * (1.f / 512.f) + 1e-5f);
    for (int c = l; c < 512; c += 32) {
      float val = (r[c] - mean) * rstd * g0[c] + b0[c];
      r[c] = val;
      O16[row * 512 + c] = (_Float16)val;
    }
  }
  __syncthreads();
  v8f acc[4] = {zero8(), zero8(), zero8(), zero8()};
  for (int k = 0; k < 512; k += 32) {
    v16h a = a_frag16(O16 + k, 512);
#pragma unroll
    for (int j = 0; j < 4; ++j) {
      int nf = wave + 8 * j;
      v16h bf = b_fragW(WoT + (size_t)(nf * 16) * 512 + k, 512);
      acc[j] = wmma16(a, bf, acc[j]);
    }
  }
  __syncthreads();
#pragma unroll
  for (int j = 0; j < 4; ++j) {
    int nf = wave + 8 * j;
#pragma unroll
    for (int r = 0; r < 8; ++r) {
      int m = r + 8 * hf, n = nf * 16 + n0;
      Obuf[m * 512 + n] += fmaxf(acc[j][r] + bo[n], 0.f);
    }
  }
  __syncthreads();
  for (int row = wave; row < 16; row += 8) {
    float* r = Obuf + row * 512;
    float s = 0.f;
    for (int c = l; c < 512; c += 32) s += r[c];
#pragma unroll
    for (int off = 16; off > 0; off >>= 1) s += __shfl_xor(s, off, 32);
    float mean = s * (1.f / 512.f);
    float v = 0.f;
    for (int c = l; c < 512; c += 32) { float d = r[c] - mean; v += d * d; }
#pragma unroll
    for (int off = 16; off > 0; off >>= 1) v += __shfl_xor(v, off, 32);
    float rstd = rsqrtf(v * (1.f / 512.f) + 1e-5f);
    if (row < nrows)
      for (int c = l; c < 512; c += 32)
        dst[(size_t)row * dst_stride + c] = (r[c] - mean) * rstd * g1[c] + b1[c];
  }
}

__global__ void post_kernel(const float* __restrict__ src, float* __restrict__ dst,
                            const _Float16* __restrict__ WoT, const float* __restrict__ bo,
                            const float* __restrict__ g0, const float* __restrict__ b0,
                            const float* __restrict__ g1, const float* __restrict__ b1,
                            int nrows_total) {
  __shared__ float Obuf[16 * 512];
  __shared__ _Float16 O16[16 * 512];
  int row0 = blockIdx.x * 16;
  for (int i = threadIdx.x; i < 16 * 512; i += 256) {
    int m = i >> 9, c = i & 511;
    int gr = row0 + m;
    Obuf[i] = (gr < nrows_total) ? src[(size_t)gr * 512 + c] : 0.f;
  }
  __syncthreads();
  int nrows = nrows_total - row0; if (nrows > 16) nrows = 16;
  post_block(Obuf, O16, WoT, bo, g0, b0, g1, b1, dst + (size_t)row0 * 512, 512, nrows);
}

// ---------------------------------------------------------------------------
// 4) MAB1 fused: per 16-row x tile, attend to 64 keys (kH16/vT16 f16), then
//    LN/FFN/LN in place.  grid = B*256.  xt doubles as the post-block f16 scratch.
// ---------------------------------------------------------------------------
__global__ void mab1_kernel(float* __restrict__ x,
                            const _Float16* __restrict__ kH16, const _Float16* __restrict__ vT16,
                            const _Float16* __restrict__ WqT, const float* __restrict__ bq,
                            const _Float16* __restrict__ WoT, const float* __restrict__ bo,
                            const float* __restrict__ g0, const float* __restrict__ b0,
                            const float* __restrict__ g1, const float* __restrict__ b1) {
  int b = blockIdx.x >> 8;
  int tile = blockIdx.x & 255;
  float* xrow = x + ((size_t)b * 4096 + (size_t)tile * 16) * 512;
  const _Float16* kHb = kH16 + (size_t)b * 64 * 512;   // [key][feat]
  const _Float16* vTb = vT16 + (size_t)b * 512 * 64;   // [feat][key]
  int wave = threadIdx.x >> 5, l = threadIdx.x & 31, n0 = l & 15, hf = l >> 4;

  __shared__ _Float16 xt[16 * 512];   // 16 KB (x tile; later post-block f16 scratch)
  __shared__ float Obuf[16 * 512];    // 32 KB
  __shared__ float qbuf[16 * 64];     // 4 KB  (q f32, residual)
  __shared__ _Float16 qb16[16 * 64];  // 2 KB  (q f16, scores A side)
  __shared__ float sbuf[16 * 64];     // 4 KB

  unsigned* xtu = (unsigned*)xt;
  for (int i4 = threadIdx.x; i4 < 16 * 128; i4 += 256) {
    int m = i4 >> 7, c4 = (i4 & 127) << 2;
    float4 vv = *(const float4*)(xrow + (size_t)m * 512 + c4);
    int o = (m * 512 + c4) >> 1;
    xtu[o] = pack2h(vv.x, vv.y);
    xtu[o + 1] = pack2h(vv.z, vv.w);
  }
  __syncthreads();

  for (int h = 0; h < 8; ++h) {
    // q_h (16x64)
    for (int f = wave; f < 4; f += 8) {
      v8f acc = zero8();
      for (int k = 0; k < 512; k += 32) {
        v16h a  = a_frag16(xt + k, 512);
        v16h bf = b_fragW(WqT + (size_t)(h * 64 + f * 16) * 512 + k, 512);
        acc = wmma16(a, bf, acc);
      }
#pragma unroll
      for (int r = 0; r < 8; ++r) {
        int m = r + 8 * hf, n = f * 16 + n0;
        float val = acc[r] + bq[h * 64 + n];
        qbuf[m * 64 + n] = val;
        qb16[m * 64 + n] = (_Float16)val;
      }
    }
    __syncthreads();
    // scores (16x64) = q_h @ kH_h^T * scale
    for (int f = wave; f < 4; f += 8) {
      v8f acc = zero8();
#pragma unroll
      for (int kk = 0; kk < 64; kk += 32) {
        v16h a  = a_frag16(qb16 + kk, 64);
        v16h bf = b_fragW(kHb + (size_t)(f * 16) * 512 + h * 64 + kk, 512);
        acc = wmma16(a, bf, acc);
      }
#pragma unroll
      for (int r = 0; r < 8; ++r)
        sbuf[(r + 8 * hf) * 64 + f * 16 + n0] = acc[r] * SCALE_QK;
    }
    __syncthreads();
    // softmax (64 keys); write P f16 in place (row stride 128 halves)
    for (int row = wave; row < 16; row += 8) {
      float* srow = sbuf + row * 64;
      float v0 = srow[l], v1 = srow[l + 32];
      float mx = fmaxf(v0, v1);
#pragma unroll
      for (int off = 16; off > 0; off >>= 1) mx = fmaxf(mx, __shfl_xor(mx, off, 32));
      float p0 = __expf(v0 - mx), p1 = __expf(v1 - mx);
      float s = p0 + p1;
#pragma unroll
      for (int off = 16; off > 0; off >>= 1) s += __shfl_xor(s, off, 32);
      float inv = 1.f / s;
      _Float16* prow = (_Float16*)srow;
      prow[l] = (_Float16)(p0 * inv); prow[l + 32] = (_Float16)(p1 * inv);
    }
    __syncthreads();
    // O_h = q_h + P @ vH_h
    const _Float16* psb = (const _Float16*)sbuf;     // stride 128 halves
    for (int f = wave; f < 4; f += 8) {
      v8f acc = zero8();
#pragma unroll
      for (int kk = 0; kk < 64; kk += 32) {
        v16h a  = a_frag16(psb + kk, 128);
        v16h bf = b_fragW(vTb + (size_t)(h * 64 + f * 16) * 64 + kk, 64);
        acc = wmma16(a, bf, acc);
      }
#pragma unroll
      for (int r = 0; r < 8; ++r) {
        int m = r + 8 * hf, n = f * 16 + n0;
        Obuf[m * 512 + h * 64 + n] = qbuf[m * 64 + n] + acc[r];
      }
    }
    __syncthreads();
  }
  // xt is dead now; reuse it as the f16 LN0 mirror for the FFN
  post_block(Obuf, xt, WoT, bo, g0, b0, g1, b1, xrow, 512, 16);
}

// ---------------------------------------------------------------------------
// 5) Final head (32 rows): MLP + LN + relu + MLP + L2 normalize.  grid = B.
// ---------------------------------------------------------------------------
__global__ void final_kernel(const float* __restrict__ pooled,
                             const float* __restrict__ Wp1, const float* __restrict__ bp1,
                             const float* __restrict__ g, const float* __restrict__ bln,
                             const float* __restrict__ Wp2, const float* __restrict__ bp2,
                             float* __restrict__ out) {
  int b = blockIdx.x, t = threadIdx.x;
  __shared__ float xs[512], hs[512], os[512], red[8];
  for (int c = t; c < 512; c += 256) xs[c] = pooled[(size_t)b * 512 + c];
  __syncthreads();
  for (int c = t; c < 512; c += 256) {
    float acc = bp1[c];
    for (int k = 0; k < 512; ++k) acc += xs[k] * Wp1[(size_t)k * 512 + c];
    hs[c] = acc;
  }
  __syncthreads();
  float s = 0.f;
  for (int c = t; c < 512; c += 256) s += hs[c];
#pragma unroll
  for (int off = 16; off > 0; off >>= 1) s += __shfl_xor(s, off, 32);
  if ((t & 31) == 0) red[t >> 5] = s;
  __syncthreads();
  float mean = 0.f;
  for (int w = 0; w < 8; ++w) mean += red[w];
  mean *= (1.f / 512.f);
  __syncthreads();
  float v = 0.f;
  for (int c = t; c < 512; c += 256) { float d = hs[c] - mean; v += d * d; }
#pragma unroll
  for (int off = 16; off > 0; off >>= 1) v += __shfl_xor(v, off, 32);
  if ((t & 31) == 0) red[t >> 5] = v;
  __syncthreads();
  float var = 0.f;
  for (int w = 0; w < 8; ++w) var += red[w];
  float rstd = rsqrtf(var * (1.f / 512.f) + 1e-5f);
  __syncthreads();
  for (int c = t; c < 512; c += 256) hs[c] = fmaxf((hs[c] - mean) * rstd * g[c] + bln[c], 0.f);
  __syncthreads();
  float ss = 0.f;
  for (int c = t; c < 512; c += 256) {
    float acc = bp2[c];
    for (int k = 0; k < 512; ++k) acc += hs[k] * Wp2[(size_t)k * 512 + c];
    os[c] = acc; ss += acc * acc;
  }
#pragma unroll
  for (int off = 16; off > 0; off >>= 1) ss += __shfl_xor(ss, off, 32);
  if ((t & 31) == 0) red[t >> 5] = ss;
  __syncthreads();
  float tot = 0.f;
  for (int w = 0; w < 8; ++w) tot += red[w];
  float inv = 1.f / fmaxf(sqrtf(tot), 1e-6f);
  for (int c = t; c < 512; c += 256) out[(size_t)b * 512 + c] = os[c] * inv;
}

// ---------------------------------------------------------------------------
// Host launcher
// ---------------------------------------------------------------------------
struct MabP {
  const float *Wq, *bq, *Wk, *bk, *Wv, *bv, *Wo, *bo, *g0, *b0, *g1, *b1;
};
static MabP get_mab(void* const* d_in, int base) {
  MabP p;
  p.Wq = (const float*)d_in[base + 0];  p.bq = (const float*)d_in[base + 1];
  p.Wk = (const float*)d_in[base + 2];  p.bk = (const float*)d_in[base + 3];
  p.Wv = (const float*)d_in[base + 4];  p.bv = (const float*)d_in[base + 5];
  p.Wo = (const float*)d_in[base + 6];  p.bo = (const float*)d_in[base + 7];
  p.g0 = (const float*)d_in[base + 8];  p.b0 = (const float*)d_in[base + 9];
  p.g1 = (const float*)d_in[base + 10]; p.b1 = (const float*)d_in[base + 11];
  return p;
}

extern "C" void kernel_launch(void* const* d_in, const int* in_sizes, int n_in,
                              void* d_out, int out_size, void* d_ws, size_t ws_size,
                              hipStream_t stream) {
  (void)in_sizes; (void)n_in; (void)out_size; (void)ws_size;
  const int B = 32, N = 4096;

  const float* xv  = (const float*)d_in[0];
  const float* res = (const float*)d_in[1];
  const float* Wi  = (const float*)d_in[2];
  const float* bi  = (const float*)d_in[3];

  char* base = (char*)d_ws;
  float* x      = (float*)base;  base += (size_t)B * N * 512 * 4;       // 268 MB
  float* HiPre  = (float*)base;  base += (size_t)B * 64 * 512 * 4;
  float* Hi     = (float*)base;  base += (size_t)B * 64 * 512 * 4;
  float* qbuf   = (float*)base;  base += (size_t)64 * 512 * 4;
  float* pooledPre = (float*)base; base += (size_t)B * 512 * 4;
  float* pooled    = (float*)base; base += (size_t)B * 512 * 4;
  _Float16* kH16 = (_Float16*)base; base += (size_t)B * 64 * 512 * 2;
  _Float16* vT16 = (_Float16*)base; base += (size_t)B * 512 * 64 * 2;
  _Float16* WTpool = (_Float16*)base; base += (size_t)28 * 512 * 512 * 2;

  auto WT = [&](int i) { return WTpool + (size_t)i * 512 * 512; };

  // ---- weight pre-pass: 8 per layer + 4 pma
  for (int lyr = 0; lyr < 3; ++lyr) {
    int ib = 4 + 25 * lyr;
    const float* Ws[8] = {
      (const float*)d_in[ib + 1],  (const float*)d_in[ib + 3],
      (const float*)d_in[ib + 5],  (const float*)d_in[ib + 7],   // Wq0 Wk0 Wv0 Wo0
      (const float*)d_in[ib + 13], (const float*)d_in[ib + 15],
      (const float*)d_in[ib + 17], (const float*)d_in[ib + 19]}; // Wq1 Wk1 Wv1 Wo1
    for (int j = 0; j < 8; ++j)
      wtrans_kernel<<<256, 256, 0, stream>>>(Ws[j], WT(lyr * 8 + j));
  }
  {
    const float* Ws[4] = {(const float*)d_in[80], (const float*)d_in[82],
                          (const float*)d_in[84], (const float*)d_in[86]};
    for (int j = 0; j < 4; ++j)
      wtrans_kernel<<<256, 256, 0, stream>>>(Ws[j], WT(24 + j));
  }

  embed_kernel<<<B * N, 256, 0, stream>>>(xv, res, Wi, bi, x);

  for (int lyr = 0; lyr < 3; ++lyr) {
    int ib = 4 + 25 * lyr;
    const float* I = (const float*)d_in[ib];
    MabP m0 = get_mab(d_in, ib + 1);
    MabP m1 = get_mab(d_in, ib + 13);
    const _Float16 *Wq0T = WT(lyr * 8 + 0), *Wk0T = WT(lyr * 8 + 1),
                   *Wv0T = WT(lyr * 8 + 2), *Wo0T = WT(lyr * 8 + 3),
                   *Wq1T = WT(lyr * 8 + 4), *Wk1T = WT(lyr * 8 + 5),
                   *Wv1T = WT(lyr * 8 + 6), *Wo1T = WT(lyr * 8 + 7);
    gemm_bias_512<0><<<4, 256, 0, stream>>>(I, 512, Wq0T, m0.bq, qbuf, 512, 64);
    isab_attn<64><<<B * 8, 256, 0, stream>>>(x, qbuf, Wk0T, m0.bk, Wv0T, m0.bv, HiPre, 64);
    post_kernel<<<B * 64 / 16, 256, 0, stream>>>(HiPre, Hi, Wo0T, m0.bo,
                                                 m0.g0, m0.b0, m0.g1, m0.b1, B * 64);
    gemm_bias_512<1><<<B * 64 / 16, 256, 0, stream>>>(Hi, 512, Wk1T, m1.bk, kH16, 512, B * 64);
    gemm_bias_512<2><<<B * 64 / 16, 256, 0, stream>>>(Hi, 512, Wv1T, m1.bv, vT16, 512, B * 64);
    mab1_kernel<<<B * 256, 256, 0, stream>>>(x, kH16, vT16, Wq1T, m1.bq, Wo1T, m1.bo,
                                             m1.g0, m1.b0, m1.g1, m1.b1);
  }

  const float* S = (const float*)d_in[79];
  MabP pm = get_mab(d_in, 80);
  const _Float16 *WqPT = WT(24), *WkPT = WT(25), *WvPT = WT(26), *WoPT = WT(27);
  gemm_bias_512<0><<<1, 256, 0, stream>>>(S, 512, WqPT, pm.bq, qbuf, 512, 1);
  isab_attn<16><<<B * 8, 256, 0, stream>>>(x, qbuf, WkPT, pm.bk, WvPT, pm.bv, pooledPre, 1);
  post_kernel<<<B / 16, 256, 0, stream>>>(pooledPre, pooled, WoPT, pm.bo,
                                          pm.g0, pm.b0, pm.g1, pm.b1, B);

  final_kernel<<<B, 256, 0, stream>>>(pooled,
      (const float*)d_in[92], (const float*)d_in[93],
      (const float*)d_in[94], (const float*)d_in[95],
      (const float*)d_in[96], (const float*)d_in[97],
      (float*)d_out);
}
